// MultiHeadAttention_68822555951114
// MI455X (gfx1250) — matrix operand order repaired and checked by reading, not compiled, not run on hardware
//
#include <hip/hip_runtime.h>

// ---------------------------------------------------------------------------
// Types for CDNA5 WMMA (wave32, 16x16x32 bf16 -> f32)
// ---------------------------------------------------------------------------
typedef __attribute__((ext_vector_type(16))) __bf16         v16bf;
typedef __attribute__((ext_vector_type(8)))  float          v8f;
typedef __attribute__((ext_vector_type(8)))  unsigned short ushort8; // 16 bytes
typedef __attribute__((ext_vector_type(4)))  unsigned short ushort4v;

union BFrag {
    v16bf bf;
    struct { ushort8 lo, hi; } u;
};

__device__ __forceinline__ v8f wmma_bf16(v16bf a, v16bf b, v8f c) {
    return __builtin_amdgcn_wmma_f32_16x16x32_bf16(
        /*neg_a=*/false, a, /*neg_b=*/false, b,
        /*c_mod=*/(short)0, c, /*reuse_a=*/false, /*reuse_b=*/false);
}

__device__ __forceinline__ unsigned short f2bf(float f) {
    unsigned int u = __float_as_uint(f);
    u += 0x7fffu + ((u >> 16) & 1u);     // round-to-nearest-even
    return (unsigned short)(u >> 16);
}

// ---------------------------------------------------------------------------
// Async global -> LDS staging (gfx1250 ASYNCcnt path), with safe fallback
// ---------------------------------------------------------------------------
#if defined(__has_builtin)
#if __has_builtin(__builtin_amdgcn_global_load_async_to_lds_b128)
#define HAVE_ASYNC_LDS 1
#endif
#endif
#ifndef HAVE_ASYNC_LDS
#define HAVE_ASYNC_LDS 0
#endif

#if HAVE_ASYNC_LDS
// Builtin prototype (from hipcc diagnostic): pointee is GCC-vector int4,
// param 1 in the global (__device__ / AS1) space, param 2 in LDS (AS3).
typedef int v4i_vs __attribute__((__vector_size__(16)));
typedef __attribute__((address_space(1))) v4i_vs GAS_v4i;
typedef __attribute__((address_space(3))) v4i_vs LAS_v4i;
#endif

__device__ __forceinline__ void async_copy_b128(const unsigned short* g, unsigned short* l) {
#if HAVE_ASYNC_LDS
    // flat LDS address: low 32 bits are the LDS offset (ISA 10.2 aperture rules)
    __builtin_amdgcn_global_load_async_to_lds_b128(
        (GAS_v4i*)(unsigned long long)g,
        (LAS_v4i*)(unsigned int)(unsigned long long)l, 0, 0);
#else
    *(ushort8*)l = *(const ushort8*)g;
#endif
}

__device__ __forceinline__ void async_wait_all() {
#if HAVE_ASYNC_LDS
#if __has_builtin(__builtin_amdgcn_s_wait_asynccnt)
    __builtin_amdgcn_s_wait_asynccnt(0);
#else
    asm volatile("s_wait_asynccnt 0x0" ::: "memory");
#endif
#endif
}

// A fragment (16x32, bf16), source row-major [M][ldK].
// Per ISA: lanes 0-15: row=lane, K=0..7 (V0-3), K=16..23 (V4-7)
//          lanes 16-31: row=lane-16, K=8..15 (V0-3), K=24..31 (V4-7)
__device__ __forceinline__ v16bf load_frag_a(const unsigned short* base, int ld, int lane) {
    const int m  = lane & 15;
    const int kb = (lane >> 4) << 3;            // 0 or 8
    const unsigned short* p = base + (size_t)m * ld + kb;
    BFrag f;
    f.u.lo = *(const ushort8*)(p);
    f.u.hi = *(const ushort8*)(p + 16);
    return f.bf;
}

// B fragment (32x16, bf16), source stored TRANSPOSED row-major [N][ldK] (Bt[n][k]).
// Same K order as A: lanes 0-15: col=lane, K=0..7, K=16..23; lanes 16-31: K=8..15, K=24..31.
__device__ __forceinline__ v16bf load_frag_b(const unsigned short* base, int ld, int lane) {
    const int n  = lane & 15;
    const int kb = (lane >> 4) << 3;            // 0 or 8
    const unsigned short* p = base + (size_t)n * ld + kb;
    BFrag f;
    f.u.lo = *(const ushort8*)(p);
    f.u.hi = *(const ushort8*)(p + 16);
    return f.bf;
}

// ---------------------------------------------------------------------------
// Problem constants
// ---------------------------------------------------------------------------
#define BATCH   4
#define SEQ     2048
#define DMODEL  1024
#define NHEADS  16
#define HDIM    64
#define MTOT    (BATCH * SEQ)        // 8192

// ---------------------------------------------------------------------------
// Kernel 1: fp32 -> bf16 convert (x), 4 elems/thread
// ---------------------------------------------------------------------------
__global__ __launch_bounds__(256) void k_cvt_bf16(const float* __restrict__ src,
                                                  unsigned short* __restrict__ dst) {
    const size_t i = ((size_t)blockIdx.x * 256 + threadIdx.x) * 4;
    float4 v = *(const float4*)(src + i);
    ushort4v o;
    o.x = f2bf(v.x); o.y = f2bf(v.y); o.z = f2bf(v.z); o.w = f2bf(v.w);
    *(ushort4v*)(dst + i) = o;
}

// ---------------------------------------------------------------------------
// Kernel 2: transpose + convert weights: src fp32 [K][N] -> dst bf16 [N][K]
// block (32,8), grid (N/32, K/32)
// ---------------------------------------------------------------------------
__global__ __launch_bounds__(256) void k_transpose_cvt(const float* __restrict__ src,
                                                       unsigned short* __restrict__ dst,
                                                       int K, int N) {
    __shared__ float tile[32][33];
    const int n0 = blockIdx.x * 32;
    const int k0 = blockIdx.y * 32;
    const int tx = threadIdx.x, ty = threadIdx.y;
#pragma unroll
    for (int i = 0; i < 32; i += 8)
        tile[ty + i][tx] = src[(size_t)(k0 + ty + i) * N + (n0 + tx)];
    __syncthreads();
#pragma unroll
    for (int i = 0; i < 32; i += 8)
        dst[(size_t)(n0 + ty + i) * K + (k0 + tx)] = f2bf(tile[tx][ty + i]);
}

// ---------------------------------------------------------------------------
// Kernel 3: QKV GEMM.  C[8192,3072] = Xbf[8192,1024] * WqkvT^T + b_qkv
// Wave tile 32x32 (4 WMMA / K-step).  Block = 8 waves as 2(M) x 4(N) = 64x128.
// Epilogue scatters: Q,K -> [BH][T][64] bf16 ; V -> [BH][64][T] bf16 (transposed).
// ---------------------------------------------------------------------------
__global__ __launch_bounds__(256) void k_gemm_qkv(const unsigned short* __restrict__ xbf,
                                                  const unsigned short* __restrict__ wt,  // [3072][1024]
                                                  const float* __restrict__ bias,         // [3072]
                                                  unsigned short* __restrict__ qb,
                                                  unsigned short* __restrict__ kb,
                                                  unsigned short* __restrict__ vtb) {
    const int lane = threadIdx.x & 31;
    const int wave = threadIdx.x >> 5;
    const int m0 = blockIdx.x * 64  + (wave >> 2) * 32;
    const int n0 = blockIdx.y * 128 + (wave & 3) * 32;
    const int K  = DMODEL;

    const unsigned short* aBase = xbf + (size_t)m0 * K;
    const unsigned short* bBase = wt  + (size_t)n0 * K;

    v8f c00 = {}, c01 = {}, c10 = {}, c11 = {};
    for (int k0 = 0; k0 < K; k0 += 32) {
        __builtin_prefetch(aBase + k0 + 256, 0, 1);
        __builtin_prefetch(bBase + k0 + 256, 0, 1);
        v16bf a0 = load_frag_a(aBase + k0,                K, lane);
        v16bf a1 = load_frag_a(aBase + (size_t)16*K + k0, K, lane);
        v16bf b0 = load_frag_b(bBase + k0,                K, lane);
        v16bf b1 = load_frag_b(bBase + (size_t)16*K + k0, K, lane);
        c00 = wmma_bf16(a0, b0, c00);
        c01 = wmma_bf16(a0, b1, c01);
        c10 = wmma_bf16(a1, b0, c10);
        c11 = wmma_bf16(a1, b1, c11);
    }

    const int coll = lane & 15;
    const int rhalf = (lane >> 4) * 8;
    const float bias0 = bias[n0 + coll];
    const float bias1 = bias[n0 + 16 + coll];

#pragma unroll
    for (int tm = 0; tm < 2; tm++) {
#pragma unroll
        for (int tn = 0; tn < 2; tn++) {
            const v8f cc = tm == 0 ? (tn == 0 ? c00 : c01) : (tn == 0 ? c10 : c11);
            const int   col   = n0 + tn * 16 + coll;
            const float bval  = tn == 0 ? bias0 : bias1;
            const int which = col >> 10;          // 0=q 1=k 2=v
            const int h     = (col >> 6) & 15;
            const int d     = col & 63;
#pragma unroll
            for (int r = 0; r < 8; r++) {
                const int row = m0 + tm * 16 + r + rhalf;   // b*T + t
                const int b = row >> 11, t = row & (SEQ - 1);
                const int bh = b * NHEADS + h;
                const unsigned short val = f2bf(cc[r] + bval);
                if (which == 0)      qb [((size_t)bh * SEQ + t) * HDIM + d] = val;
                else if (which == 1) kb [((size_t)bh * SEQ + t) * HDIM + d] = val;
                else                 vtb[((size_t)bh * HDIM + d) * SEQ + t] = val;
            }
        }
    }
}

// ---------------------------------------------------------------------------
// Kernel 4: flash attention, LDS-shared K/V with async double buffering.
// Block = 8 waves, all on the same (b,h); wave w owns q rows [q0+16w, q0+16w+16).
// Per 32-key block: stage K (32x64) and V (64x32) bf16 tiles into LDS with
// global_load_async_to_lds_b128 (ping-pong), 4 score WMMAs, online softmax,
// P relayout via per-wave LDS slice, 4 O-update WMMAs.
// ---------------------------------------------------------------------------
__global__ __launch_bounds__(256) void k_attention(const unsigned short* __restrict__ qb,
                                                   const unsigned short* __restrict__ kbuf,
                                                   const unsigned short* __restrict__ vtb,
                                                   unsigned short* __restrict__ attnbf) {
    __shared__ __align__(16) unsigned short Kt[2][32 * 64];   // 2 x 4KB
    __shared__ __align__(16) unsigned short Vt_s[2][64 * 32]; // 2 x 4KB
    __shared__ __align__(16) unsigned short Pt[8][16 * 32];   // 8 x 1KB

    const int tid  = threadIdx.x;
    const int lane = tid & 31;
    const int wave = tid >> 5;
    const int bh   = blockIdx.y;                 // 0..63
    const int q0   = blockIdx.x * 128 + wave * 16;

    const unsigned short* Q  = qb   + ((size_t)bh * SEQ + q0) * HDIM;
    const unsigned short* Kp = kbuf + (size_t)bh * SEQ * HDIM;
    const unsigned short* Vt = vtb  + (size_t)bh * HDIM * SEQ;

    // staging indices: K tile 32 rows x 8 b128 segs; V tile 64 rows x 4 segs
    const int krow = tid >> 3, kseg = (tid & 7) * 8;
    const int vrow = tid >> 2, vseg = (tid & 3) * 8;

    const v16bf qf0 = load_frag_a(Q,      HDIM, lane);  // d 0..31
    const v16bf qf1 = load_frag_a(Q + 32, HDIM, lane);  // d 32..63

    v8f o0 = {}, o1 = {}, o2 = {}, o3 = {};
    float mrow[8], lrow[8];
#pragma unroll
    for (int r = 0; r < 8; r++) { mrow[r] = -1e30f; lrow[r] = 0.0f; }

    unsigned short* L = Pt[wave];
    const int half = lane >> 4;
    const int coll = lane & 15;
    const float scale = 0.125f;                  // 1/sqrt(64)

    // prologue: stage key block 0 into buffer 0
    async_copy_b128(Kp + (size_t)krow * HDIM + kseg, Kt[0] + krow * 64 + kseg);
    async_copy_b128(Vt + (size_t)vrow * SEQ  + vseg, Vt_s[0] + vrow * 32 + vseg);
    async_wait_all();
    __syncthreads();

    for (int key0 = 0; key0 < SEQ; key0 += 32) {
        const int cur = (key0 >> 5) & 1;
        const int nxt = cur ^ 1;
        // prefetch next key block into the idle buffer (overlaps with compute)
        if (key0 + 32 < SEQ) {
            async_copy_b128(Kp + (size_t)(key0 + 32 + krow) * HDIM + kseg,
                            Kt[nxt] + krow * 64 + kseg);
            async_copy_b128(Vt + (size_t)vrow * SEQ + (key0 + 32) + vseg,
                            Vt_s[nxt] + vrow * 32 + vseg);
        }

        // ---- scores: two 16-key tiles, K=64 accumulated over two 32-chunks
        const unsigned short* Kl = Kt[cur];
        v16bf kA0 = load_frag_b(Kl,                64, lane);
        v16bf kA1 = load_frag_b(Kl + 32,           64, lane);
        v16bf kB0 = load_frag_b(Kl + 16 * 64,      64, lane);
        v16bf kB1 = load_frag_b(Kl + 16 * 64 + 32, 64, lane);
        v8f s0 = {}, s1 = {};
        s0 = wmma_bf16(qf0, kA0, s0);  s0 = wmma_bf16(qf1, kA1, s0);
        s1 = wmma_bf16(qf0, kB0, s1);  s1 = wmma_bf16(qf1, kB1, s1);

        // ---- online softmax (row r -> actual row r + 8*half; cols across 16 lanes)
        float alpha[8];
#pragma unroll
        for (int r = 0; r < 8; r++) {
            float v0 = s0[r] * scale, v1 = s1[r] * scale;
            float mx = fmaxf(v0, v1);
            mx = fmaxf(mx, __shfl_xor(mx, 1, 32));
            mx = fmaxf(mx, __shfl_xor(mx, 2, 32));
            mx = fmaxf(mx, __shfl_xor(mx, 4, 32));
            mx = fmaxf(mx, __shfl_xor(mx, 8, 32));
            const float mnew = fmaxf(mrow[r], mx);
            alpha[r] = __expf(mrow[r] - mnew);
            mrow[r] = mnew;
            const float p0 = __expf(v0 - mnew);
            const float p1 = __expf(v1 - mnew);
            s0[r] = p0; s1[r] = p1;
            float rs = p0 + p1;
            rs += __shfl_xor(rs, 1, 32);
            rs += __shfl_xor(rs, 2, 32);
            rs += __shfl_xor(rs, 4, 32);
            rs += __shfl_xor(rs, 8, 32);
            lrow[r] = lrow[r] * alpha[r] + rs;
        }
#pragma unroll
        for (int r = 0; r < 8; r++) {
            o0[r] *= alpha[r]; o1[r] *= alpha[r];
            o2[r] *= alpha[r]; o3[r] *= alpha[r];
        }

        // ---- P (16q x 32key) -> per-wave LDS slice, re-read as A fragment
#pragma unroll
        for (int r = 0; r < 8; r++) {
            const int row = r + 8 * half;
            L[row * 32 + coll]      = f2bf(s0[r]);
            L[row * 32 + 16 + coll] = f2bf(s1[r]);
        }
        const v16bf pf = load_frag_a(L, 32, lane);   // same-wave DS ops are in order

        // ---- O += P @ V  (K = 32 keys, N = 64 d in 4 tiles) from LDS V tile
        const unsigned short* Vl = Vt_s[cur];
        o0 = wmma_bf16(pf, load_frag_b(Vl +  0 * 32, 32, lane), o0);
        o1 = wmma_bf16(pf, load_frag_b(Vl + 16 * 32, 32, lane), o1);
        o2 = wmma_bf16(pf, load_frag_b(Vl + 32 * 32, 32, lane), o2);
        o3 = wmma_bf16(pf, load_frag_b(Vl + 48 * 32, 32, lane), o3);

        // prefetch complete + everyone done with 'cur' before it is re-used
        async_wait_all();
        __syncthreads();
    }

    // ---- finalize: divide by l, write bf16 attn [B*T][D] at cols h*64..h*64+63
    const int h = bh & 15;
    const size_t browbase = (size_t)(bh >> 4) * SEQ;
#pragma unroll
    for (int r = 0; r < 8; r++) {
        const float inv = 1.0f / lrow[r];
        const int t = q0 + r + 8 * half;
        const size_t rowoff = (browbase + t) * DMODEL + h * HDIM;
        attnbf[rowoff +  0 + coll] = f2bf(o0[r] * inv);
        attnbf[rowoff + 16 + coll] = f2bf(o1[r] * inv);
        attnbf[rowoff + 32 + coll] = f2bf(o2[r] * inv);
        attnbf[rowoff + 48 + coll] = f2bf(o3[r] * inv);
    }
}

// ---------------------------------------------------------------------------
// Kernel 5: output projection.  out[8192,1024] = attnbf @ w_outT^T + b_out (fp32)
// ---------------------------------------------------------------------------
__global__ __launch_bounds__(256) void k_gemm_out(const unsigned short* __restrict__ abf,
                                                  const unsigned short* __restrict__ wt,  // [1024][1024]
                                                  const float* __restrict__ bias,
                                                  float* __restrict__ out) {
    const int lane = threadIdx.x & 31;
    const int wave = threadIdx.x >> 5;
    const int m0 = blockIdx.x * 64  + (wave >> 2) * 32;
    const int n0 = blockIdx.y * 128 + (wave & 3) * 32;
    const int K  = DMODEL;

    const unsigned short* aBase = abf + (size_t)m0 * K;
    const unsigned short* bBase = wt  + (size_t)n0 * K;

    v8f c00 = {}, c01 = {}, c10 = {}, c11 = {};
    for (int k0 = 0; k0 < K; k0 += 32) {
        __builtin_prefetch(aBase + k0 + 256, 0, 1);
        __builtin_prefetch(bBase + k0 + 256, 0, 1);
        v16bf a0 = load_frag_a(aBase + k0,                K, lane);
        v16bf a1 = load_frag_a(aBase + (size_t)16*K + k0, K, lane);
        v16bf b0 = load_frag_b(bBase + k0,                K, lane);
        v16bf b1 = load_frag_b(bBase + (size_t)16*K + k0, K, lane);
        c00 = wmma_bf16(a0, b0, c00);
        c01 = wmma_bf16(a0, b1, c01);
        c10 = wmma_bf16(a1, b0, c10);
        c11 = wmma_bf16(a1, b1, c11);
    }

    const int coll = lane & 15;
    const int rhalf = (lane >> 4) * 8;
    const float bias0 = bias[n0 + coll];
    const float bias1 = bias[n0 + 16 + coll];
#pragma unroll
    for (int tm = 0; tm < 2; tm++) {
#pragma unroll
        for (int tn = 0; tn < 2; tn++) {
            const v8f cc = tm == 0 ? (tn == 0 ? c00 : c01) : (tn == 0 ? c10 : c11);
            const int col = n0 + tn * 16 + coll;
            const float bval = tn == 0 ? bias0 : bias1;
#pragma unroll
            for (int r = 0; r < 8; r++) {
                const int row = m0 + tm * 16 + r + rhalf;
                out[(size_t)row * DMODEL + col] = cc[r] + bval;
            }
        }
    }
}

// ---------------------------------------------------------------------------
// Host launcher
// ---------------------------------------------------------------------------
extern "C" void kernel_launch(void* const* d_in, const int* in_sizes, int n_in,
                              void* d_out, int out_size, void* d_ws, size_t ws_size,
                              hipStream_t stream) {
    const float* x      = (const float*)d_in[0];   // [4,2048,1024]
    const float* w_qkv  = (const float*)d_in[1];   // [1024,3072]
    const float* b_qkv  = (const float*)d_in[2];   // [3072]
    const float* w_out  = (const float*)d_in[3];   // [1024,1024]
    const float* b_out  = (const float*)d_in[4];   // [1024]
    float* out = (float*)d_out;

    char* ws = (char*)d_ws;
    size_t off = 0;
    unsigned short* xbf   = (unsigned short*)(ws + off); off += (size_t)MTOT * DMODEL * 2;        // 16 MB
    unsigned short* wqkvT = (unsigned short*)(ws + off); off += (size_t)3 * DMODEL * DMODEL * 2;  // 6 MB
    unsigned short* woT   = (unsigned short*)(ws + off); off += (size_t)DMODEL * DMODEL * 2;      // 2 MB
    unsigned short* qb    = (unsigned short*)(ws + off); off += (size_t)MTOT * DMODEL * 2;        // 16 MB
    unsigned short* kb    = (unsigned short*)(ws + off); off += (size_t)MTOT * DMODEL * 2;        // 16 MB
    unsigned short* vtb   = (unsigned short*)(ws + off); off += (size_t)MTOT * DMODEL * 2;        // 16 MB
    unsigned short* attnb = (unsigned short*)(ws + off); off += (size_t)MTOT * DMODEL * 2;        // 16 MB

    // 1) x -> bf16
    k_cvt_bf16<<<(MTOT * DMODEL) / (256 * 4), 256, 0, stream>>>(x, xbf);
    // 2) weights -> transposed bf16
    k_transpose_cvt<<<dim3(3 * DMODEL / 32, DMODEL / 32), dim3(32, 8), 0, stream>>>(w_qkv, wqkvT, DMODEL, 3 * DMODEL);
    k_transpose_cvt<<<dim3(DMODEL / 32, DMODEL / 32),     dim3(32, 8), 0, stream>>>(w_out, woT, DMODEL, DMODEL);
    // 3) QKV projection (scatter to Q,K row-major / V transposed)
    k_gemm_qkv<<<dim3(MTOT / 64, 3 * DMODEL / 128), 256, 0, stream>>>(xbf, wqkvT, b_qkv, qb, kb, vtb);
    // 4) flash attention (8 waves share one (b,h) and LDS-staged K/V blocks)
    k_attention<<<dim3(SEQ / 128, BATCH * NHEADS), 256, 0, stream>>>(qb, kb, vtb, attnb);
    // 5) output projection + bias -> d_out
    k_gemm_out<<<dim3(MTOT / 64, DMODEL / 128), 256, 0, stream>>>(attnb, woT, b_out, out);
}
